// ResidualGraphAttention_34402688041061
// MI455X (gfx1250) — matrix operand order repaired and checked by reading, not compiled, not run on hardware
//
#include <hip/hip_runtime.h>

// ---------------- problem constants ----------------
#define NND   100000                // nodes
#define NE    1600000               // edges (without self loops)
#define EPE   (NE + NND)            // edges + self loops = 1.7M
#define HID   64
#define HEADS 2
#define DHD   32                    // HID / HEADS
#define NL    3
#define NEG_SLOPE 0.2f
#define LN_EPS    1e-5f

// ---------------- WMMA vector types (CDNA5 wave32) ----------------
typedef __attribute__((ext_vector_type(16))) __bf16         v16bf;
typedef __attribute__((ext_vector_type(16))) unsigned short v16u;
typedef __attribute__((ext_vector_type(8)))  unsigned short v8u;
typedef __attribute__((ext_vector_type(8)))  float          v8f;

// float -> bf16 bits, round-to-nearest-even
static __device__ __forceinline__ unsigned short f2bf(float f) {
  unsigned u = __float_as_uint(f);
  u += 0x7FFFu + ((u >> 16) & 1u);
  return (unsigned short)(u >> 16);
}

// two contiguous 16B chunks, 16 elements apart (A-fragment: K {0..7}+{16..23})
static __device__ __forceinline__ v16bf load_frag_split16(const unsigned short* p) {
  v8u lo = *(const v8u*)p;
  v8u hi = *(const v8u*)(p + 16);
  v16u c = __builtin_shufflevector(lo, hi, 0,1,2,3,4,5,6,7,8,9,10,11,12,13,14,15);
  return __builtin_bit_cast(v16bf, c);
}
// 32B contiguous run (B-fragment from transposed weights: K {0..15} at fixed N)
static __device__ __forceinline__ v16bf load_frag_contig(const unsigned short* p) {
  v8u lo = *(const v8u*)p;
  v8u hi = *(const v8u*)(p + 8);
  v16u c = __builtin_shufflevector(lo, hi, 0,1,2,3,4,5,6,7,8,9,10,11,12,13,14,15);
  return __builtin_bit_cast(v16bf, c);
}

// ---------------- utility kernels ----------------
__global__ void fill_u32_kernel(unsigned* __restrict__ p, unsigned v, int n) {
  int i = blockIdx.x * blockDim.x + threadIdx.x;
  if (i < n) p[i] = v;
}

// convert+transpose weights: d[l][n][k] = bf16(s[l][k][n]); makes B-fragment loads contiguous
__global__ void cvt_bf16_t_kernel(const float* __restrict__ s, unsigned short* __restrict__ d) {
  int i = blockIdx.x * blockDim.x + threadIdx.x;     // over NL*HID*HID
  if (i >= NL * HID * HID) return;
  int l = i >> 12;
  int r = i & 4095;
  int k = r >> 6, n = r & 63;
  d[l * 4096 + n * 64 + k] = f2bf(s[i]);
}

// x = coords @ Wp + bp  (K=2, pure VALU; also produce bf16 copy for WMMA)
__global__ void proj_kernel(const float* __restrict__ coords, const float* __restrict__ Wp,
                            const float* __restrict__ bp,
                            float* __restrict__ x, unsigned short* __restrict__ xb) {
  int i = blockIdx.x * blockDim.x + threadIdx.x;
  if (i >= NND * HID) return;
  int n = i >> 6, j = i & 63;
  float v = coords[2 * n] * Wp[j] + coords[2 * n + 1] * Wp[HID + j] + bp[j];
  x[i] = v;
  xb[i] = f2bf(v);
}

// ---------------- WMMA GEMM: xl = x@Wl + bl, xr = x@Wr + br ----------------
// One wave per 16-row tile of x. 4 N-tiles x 2 K-steps x 2 weight matrices
// = 16 v_wmma_f32_16x16x32_bf16 per wave; all fragments via b128 loads.
// A as loaded here (contiguous K order, B matches): lane<16 holds M=lane, K elems {0..15}+kbase;
//                          lane>=16 holds M=lane-16, K elems {16..31}+kbase.
// B as loaded here (32x16, same K order as A): lanes0-15: N=lane, K=0..15 (+kbase); lanes16-31: K+=16.
// C/D layout (f32 16x16): VGPR r -> M=r (lanes0-15) / M=r+8 (lanes16-31), N=lane%16.
__global__ void gemm_xlxr_kernel(const unsigned short* __restrict__ xb,
                                 const unsigned short* __restrict__ Wlt,   // transposed [64][64]
                                 const unsigned short* __restrict__ Wrt,   // transposed [64][64]
                                 const float* __restrict__ bl, const float* __restrict__ br,
                                 float* __restrict__ xl, float* __restrict__ xr) {
  int wave = (int)((blockIdx.x * blockDim.x + threadIdx.x) >> 5);
  int lane = threadIdx.x & 31;
  int m0 = wave * 16;
  if (m0 >= NND) return;              // wave-uniform: EXEC stays all-ones for WMMA
  int hs  = lane >> 4;                // half-select
  int l16 = lane & 15;

  v8f accl[4], accr[4];
#pragma unroll
  for (int t = 0; t < 4; t++) {
    float vl = bl[t * 16 + l16], vr = br[t * 16 + l16];
#pragma unroll
    for (int r = 0; r < 8; r++) { accl[t][r] = vl; accr[t][r] = vr; }
  }

#pragma unroll
  for (int ks = 0; ks < 2; ks++) {
    const int kbase = ks * 32;
    v16bf a = load_frag_contig(xb + (size_t)(m0 + l16) * HID + kbase + hs * 16);
#pragma unroll
    for (int t = 0; t < 4; t++) {
      int n = t * 16 + l16;
      v16bf b = load_frag_contig(Wlt + (size_t)n * HID + kbase + hs * 16);
      accl[t] = __builtin_amdgcn_wmma_f32_16x16x32_bf16(false, a, false, b, (short)0, accl[t], false, false);
      b = load_frag_contig(Wrt + (size_t)n * HID + kbase + hs * 16);
      accr[t] = __builtin_amdgcn_wmma_f32_16x16x32_bf16(false, a, false, b, (short)0, accr[t], false, false);
    }
  }

#pragma unroll
  for (int t = 0; t < 4; t++) {
    int n = t * 16 + l16;
#pragma unroll
    for (int r = 0; r < 8; r++) {
      size_t m = (size_t)(m0 + r + hs * 8);
      xl[m * HID + n] = accl[t][r];
      xr[m * HID + n] = accr[t][r];
    }
  }
}

// ---------------- edge scoring + segment max ----------------
// float atomic max via monotone int encoding (works mixed-sign with -inf init)
static __device__ __forceinline__ void atomic_max_f32(float* addr, float val) {
  if (val >= 0.f) atomicMax((int*)addr, __float_as_int(val));
  else            atomicMin((unsigned int*)addr, __float_as_uint(val));
}

__global__ void score_kernel(const int* __restrict__ esrc, const int* __restrict__ edst,
                             const float* __restrict__ xl, const float* __restrict__ xr,
                             const float* __restrict__ att,
                             float* __restrict__ score, float* __restrict__ smax) {
  int e = blockIdx.x * blockDim.x + threadIdx.x;
  if (e >= EPE) return;
  int s, d;
  if (e < NE) { s = esrc[e]; d = edst[e]; } else { s = d = e - NE; }
  const float4* pl = (const float4*)(xl + (size_t)s * HID);
  const float4* pr = (const float4*)(xr + (size_t)d * HID);
  const float4* pa = (const float4*)att;
#pragma unroll
  for (int h = 0; h < HEADS; h++) {
    float acc = 0.f;
#pragma unroll
    for (int q = 0; q < DHD / 4; q++) {
      float4 a = pl[h * (DHD / 4) + q];
      float4 b = pr[h * (DHD / 4) + q];
      float4 w = pa[h * (DHD / 4) + q];
      float v0 = a.x + b.x; v0 = (v0 > 0.f) ? v0 : v0 * NEG_SLOPE;
      float v1 = a.y + b.y; v1 = (v1 > 0.f) ? v1 : v1 * NEG_SLOPE;
      float v2 = a.z + b.z; v2 = (v2 > 0.f) ? v2 : v2 * NEG_SLOPE;
      float v3 = a.w + b.w; v3 = (v3 > 0.f) ? v3 : v3 * NEG_SLOPE;
      acc += v0 * w.x + v1 * w.y + v2 * w.z + v3 * w.w;
    }
    score[(size_t)e * HEADS + h] = acc;
    atomic_max_f32(&smax[(size_t)d * HEADS + h], acc);
  }
}

// ex = exp(score - smax[dst]) in place; denom[dst] += ex
__global__ void softmax_num_kernel(const int* __restrict__ edst,
                                   float* __restrict__ score_ex,
                                   const float* __restrict__ smax,
                                   float* __restrict__ denom) {
  int idx = blockIdx.x * blockDim.x + threadIdx.x;    // over EPE*HEADS
  if (idx >= EPE * HEADS) return;
  int e = idx >> 1, h = idx & 1;
  int d = (e < NE) ? edst[e] : e - NE;
  float m = smax[(size_t)d * HEADS + h];
  if (isnan(m) || isinf(m)) m = 0.f;                  // jnp.where(isfinite, smax, 0)
  float ex = expf(score_ex[idx] - m);
  score_ex[idx] = ex;
  atomicAdd(&denom[(size_t)d * HEADS + h], ex);
}

// out[dst] += (ex/denom[dst]) * xl[src]  -- one thread per (edge, feature)
__global__ void agg_kernel(const int* __restrict__ esrc, const int* __restrict__ edst,
                           const float* __restrict__ ex, const float* __restrict__ denom,
                           const float* __restrict__ xl, float* __restrict__ out) {
  unsigned idx = blockIdx.x * blockDim.x + threadIdx.x;   // EPE*64 = 108.8M
  if (idx >= (unsigned)EPE * HID) return;
  int e = (int)(idx >> 6);
  int j = (int)(idx & 63);
  int h = j >> 5;
  int s, d;
  if (e < NE) { s = esrc[e]; d = edst[e]; } else { s = d = e - NE; }
  float alpha = ex[(size_t)e * HEADS + h] / (denom[(size_t)d * HEADS + h] + 1e-16f);
  atomicAdd(&out[(size_t)d * HID + j], alpha * xl[(size_t)s * HID + j]);
}

// ---------------- ELU + residual + LayerNorm + nan_to_num ----------------
// one wave per node; lane handles features {lane, lane+32}; wave-reduce via shfl_xor
__global__ void post_kernel(const float* __restrict__ acc, const float* __restrict__ bias,
                            const float* __restrict__ xres,
                            const float* __restrict__ gamma, const float* __restrict__ beta,
                            float* __restrict__ xnew, unsigned short* __restrict__ xbnew) {
  int n = blockIdx.x;
  int l = threadIdx.x;                    // 0..31
  size_t base = (size_t)n * HID;
  float t0 = acc[base + l]      + bias[l];
  float t1 = acc[base + l + 32] + bias[l + 32];
  t0 = (t0 > 0.f) ? t0 : (expf(t0) - 1.f);    // ELU
  t1 = (t1 > 0.f) ? t1 : (expf(t1) - 1.f);
  float h0 = t0 + xres[base + l];
  float h1 = t1 + xres[base + l + 32];
  float s = h0 + h1;
#pragma unroll
  for (int m = 16; m > 0; m >>= 1) s += __shfl_xor(s, m, 32);
  float mu = s * (1.f / 64.f);
  float d0 = h0 - mu, d1 = h1 - mu;
  float v = d0 * d0 + d1 * d1;
#pragma unroll
  for (int m = 16; m > 0; m >>= 1) v += __shfl_xor(v, m, 32);
  float inv = rsqrtf(v * (1.f / 64.f) + LN_EPS);
  float y0 = d0 * inv * gamma[l]      + beta[l];
  float y1 = d1 * inv * gamma[l + 32] + beta[l + 32];
  if (isnan(y0)) y0 = 0.f; else if (isinf(y0)) y0 = (y0 > 0.f) ? 1e6f : -1e6f;
  if (isnan(y1)) y1 = 0.f; else if (isinf(y1)) y1 = (y1 > 0.f) ? 1e6f : -1e6f;
  xnew[base + l] = y0;
  xnew[base + l + 32] = y1;
  if (xbnew) { xbnew[base + l] = f2bf(y0); xbnew[base + l + 32] = f2bf(y1); }
}

// ---------------- launch ----------------
extern "C" void kernel_launch(void* const* d_in, const int* in_sizes, int n_in,
                              void* d_out, int out_size, void* d_ws, size_t ws_size,
                              hipStream_t stream) {
  const float* coords = (const float*)d_in[0];
  const int*   ei     = (const int*)d_in[1];   // JAX default (x64 off) -> int32
  const float* Wp     = (const float*)d_in[2];
  const float* bp     = (const float*)d_in[3];
  const float* Wl     = (const float*)d_in[4];
  const float* bl     = (const float*)d_in[5];
  const float* Wr     = (const float*)d_in[6];
  const float* br     = (const float*)d_in[7];
  const float* att    = (const float*)d_in[8];
  const float* bias_o = (const float*)d_in[9];
  const float* gamma  = (const float*)d_in[10];
  const float* beta   = (const float*)d_in[11];
  float* out = (float*)d_out;                  // [N,64] f32; also per-layer aggregation buffer
  const int* esrc = ei;
  const int* edst = ei + NE;

  // workspace carve-up (~105 MB)
  char* w = (char*)d_ws;
  float* x     = (float*)w; w += sizeof(float) * (size_t)NND * HID;
  float* xl    = (float*)w; w += sizeof(float) * (size_t)NND * HID;
  float* xr    = (float*)w; w += sizeof(float) * (size_t)NND * HID;
  float* score = (float*)w; w += sizeof(float) * (size_t)EPE * HEADS;   // reused as ex in place
  float* smax  = (float*)w; w += sizeof(float) * (size_t)NND * HEADS;
  float* denom = (float*)w; w += sizeof(float) * (size_t)NND * HEADS;
  unsigned short* xb  = (unsigned short*)w; w += sizeof(unsigned short) * (size_t)NND * HID;
  unsigned short* Wlt = (unsigned short*)w; w += sizeof(unsigned short) * (size_t)NL * HID * HID;
  unsigned short* Wrt = (unsigned short*)w; w += sizeof(unsigned short) * (size_t)NL * HID * HID;

  const int B = 256;
  cvt_bf16_t_kernel<<<(NL * HID * HID + B - 1) / B, B, 0, stream>>>(Wl, Wlt);
  cvt_bf16_t_kernel<<<(NL * HID * HID + B - 1) / B, B, 0, stream>>>(Wr, Wrt);
  proj_kernel<<<(NND * HID + B - 1) / B, B, 0, stream>>>(coords, Wp, bp, x, xb);

  const int gemm_threads = (NND / 16) * 32;    // one wave per 16-row tile
  for (int i = 0; i < NL; i++) {
    gemm_xlxr_kernel<<<(gemm_threads + B - 1) / B, B, 0, stream>>>(
        xb, Wlt + (size_t)i * HID * HID, Wrt + (size_t)i * HID * HID,
        bl + i * HID, br + i * HID, xl, xr);
    fill_u32_kernel<<<(NND * HEADS + B - 1) / B, B, 0, stream>>>((unsigned*)smax, 0xFF800000u, NND * HEADS); // -inf
    fill_u32_kernel<<<(NND * HEADS + B - 1) / B, B, 0, stream>>>((unsigned*)denom, 0u, NND * HEADS);
    fill_u32_kernel<<<(NND * HID + B - 1) / B, B, 0, stream>>>((unsigned*)out, 0u, NND * HID);
    score_kernel<<<(EPE + B - 1) / B, B, 0, stream>>>(esrc, edst, xl, xr, att + i * HEADS * DHD, score, smax);
    softmax_num_kernel<<<(EPE * HEADS + B - 1) / B, B, 0, stream>>>(edst, score, smax, denom);
    agg_kernel<<<(unsigned)(((unsigned)EPE * HID + B - 1) / B), B, 0, stream>>>(esrc, edst, score, denom, xl, out);
    post_kernel<<<NND, 32, 0, stream>>>(out, bias_o + i * HID, x, gamma + i * HID, beta + i * HID,
                                        (i < NL - 1) ? x : out,
                                        (i < NL - 1) ? xb : (unsigned short*)nullptr);
  }
}
